// GTLayer_44487271252168
// MI455X (gfx1250) — matrix operand order repaired and checked by reading, not compiled, not run on hardware
//
#include <hip/hip_runtime.h>

typedef _Float16 half_t;
typedef __attribute__((ext_vector_type(16))) _Float16 v16h;
typedef __attribute__((ext_vector_type(8)))  _Float16 v8h;
typedef __attribute__((ext_vector_type(8)))  float    v8f;

#define NN 50000
#define NE 800000
#define DM 128

// ---------------------------------------------------------------------------
// Generic WMMA GEMM:  C[nrows x Mdim] = act(A[nrows x Kdim] @ B[Kdim x Mdim]
//                                           + bias + resid)
// Block: 256 threads (8 waves). Each block covers 128 rows x 128 cols.
// K processed in panels of 32 (v_wmma_f32_16x16x32_f16, fp32 accumulate).
// ---------------------------------------------------------------------------
__global__ __launch_bounds__(256)
void gemm_wmma_kernel(const float* __restrict__ A, const float* __restrict__ B,
                      float* __restrict__ C, const float* __restrict__ bias,
                      const float* __restrict__ resid, int nrows, int Kdim,
                      int Mdim, int do_relu)
{
  __shared__ half_t sA[128 * 32];   // [row][k]  (row-major, f16)
  __shared__ half_t sBt[128 * 32];  // [col][k]  (transposed, f16)

  const int tid  = threadIdx.x;
  const int wave = tid >> 5;
  const int lane = tid & 31;
  const int row0 = blockIdx.x * 128;
  const int col0 = blockIdx.y * 128;
  const int m    = lane & 15;
  const int hi   = lane >> 4;

  v8f acc[8] = {};

  const int kPanels = Kdim >> 5;
  for (int kb = 0; kb < kPanels; ++kb) {
    // stage A panel: 128 rows x 32 k (fp32 -> f16), zero-pad past nrows
    for (int e = tid; e < 1024; e += 256) {
      int r  = e >> 3;
      int c4 = (e & 7) << 2;
      float4 v = make_float4(0.f, 0.f, 0.f, 0.f);
      int gr = row0 + r;
      if (gr < nrows)
        v = *(const float4*)(A + (size_t)gr * Kdim + (kb << 5) + c4);
      half_t* d = &sA[r * 32 + c4];
      d[0] = (half_t)v.x; d[1] = (half_t)v.y;
      d[2] = (half_t)v.z; d[3] = (half_t)v.w;
    }
    // stage B panel: 32 k x 128 cols, transposed into [col][k]
    for (int e = tid; e < 1024; e += 256) {
      int r  = e >> 5;          // k-row within panel
      int c4 = (e & 31) << 2;   // local column
      float4 v = *(const float4*)(B + (size_t)((kb << 5) + r) * Mdim + col0 + c4);
      sBt[(c4 + 0) * 32 + r] = (half_t)v.x;
      sBt[(c4 + 1) * 32 + r] = (half_t)v.y;
      sBt[(c4 + 2) * 32 + r] = (half_t)v.z;
      sBt[(c4 + 3) * 32 + r] = (half_t)v.w;
    }
    __syncthreads();

    // A fragment, ISA 16-bit 16x32 layout:
    //   lane<16: halves 0..7 -> K 0..7,  halves 8..15 -> K 16..23
    //   lane>=16: +8 on both ranges
    const half_t* ap = &sA[(wave * 16 + m) * 32];
    v8h alo = *(const v8h*)(ap + 8 * hi);
    v8h ahi = *(const v8h*)(ap + 16 + 8 * hi);
    v16h afrag = __builtin_shufflevector(alo, ahi,
        0, 1, 2, 3, 4, 5, 6, 7, 8, 9, 10, 11, 12, 13, 14, 15);

#pragma unroll
    for (int t = 0; t < 8; ++t) {
      // B fragment: same K layout as A (lane<16: K 0..7,16..23; lane>=16: +8)
      const half_t* bp = &sBt[(t * 16 + m) * 32];
      v8h blo = *(const v8h*)(bp + 8 * hi);
      v8h bhi = *(const v8h*)(bp + 16 + 8 * hi);
      v16h bfrag = __builtin_shufflevector(blo, bhi,
          0, 1, 2, 3, 4, 5, 6, 7, 8, 9, 10, 11, 12, 13, 14, 15);
      acc[t] = __builtin_amdgcn_wmma_f32_16x16x32_f16(
          false, afrag, false, bfrag, (short)0, acc[t], false, false);
    }
    __syncthreads();
  }

  // epilogue: C/D layout -> reg r holds row (r + 8*hi), col = lane&15
#pragma unroll
  for (int t = 0; t < 8; ++t) {
    int col = col0 + t * 16 + m;
    float bv = bias ? bias[col] : 0.f;
#pragma unroll
    for (int r = 0; r < 8; ++r) {
      int grow = row0 + wave * 16 + r + 8 * hi;
      if (grow < nrows) {
        float v = acc[t][r] + bv;
        if (resid) v += resid[(size_t)grow * Mdim + col];
        if (do_relu) v = fmaxf(v, 0.f);
        C[(size_t)grow * Mdim + col] = v;
      }
    }
  }
}

// ---------------------------------------------------------------------------
// Edge-scatter attention. 128 threads = 128 channels (head h = t/16, d = t%16).
// score_h = exp(clip(sum_d K[src]*Q[dst]*0.25*E[h,d], -5, 5))
// wV[dst] += V[src]*score ; Z[dst,h] += score   (global f32 atomics -> L2)
// ---------------------------------------------------------------------------
__global__ __launch_bounds__(128)
void edge_attn_kernel(const float* __restrict__ Qm, const float* __restrict__ Km,
                      const float* __restrict__ Vm,
                      const float* __restrict__ edge_attr,
                      const int* __restrict__ edge_index,
                      const float* __restrict__ We,
                      float* __restrict__ wV, float* __restrict__ Z)
{
  const int t = threadIdx.x;     // channel 0..127
  const int h = t >> 4;
  const int d = t & 15;
  const float we = We[t];

  for (int e = blockIdx.x; e < NE; e += gridDim.x) {
    int   src = edge_index[e];
    int   dst = edge_index[NE + e];
    float ea  = edge_attr[e];

    const float* vrow = Vm + (size_t)src * DM + t;
    __builtin_prefetch(vrow, 0, 0);   // global_prefetch_b8

    float kq = Km[(size_t)src * DM + t] * Qm[(size_t)dst * DM + t]
               * 0.25f * (ea * we);
    float s = kq;
    s += __shfl_xor(s, 1, 32);
    s += __shfl_xor(s, 2, 32);
    s += __shfl_xor(s, 4, 32);
    s += __shfl_xor(s, 8, 32);
    s = fminf(5.f, fmaxf(-5.f, s));
    float score = expf(s);

    float msg = *vrow * score;
    atomicAdd(&wV[(size_t)dst * DM + t], msg);
    if (d == 0) atomicAdd(&Z[(size_t)dst * 8 + h], score);
  }
}

// ---------------------------------------------------------------------------
// elementwise helpers
// ---------------------------------------------------------------------------
__global__ __launch_bounds__(256)
void zero_kernel(float* __restrict__ p, size_t n)
{
  for (size_t i = (size_t)blockIdx.x * blockDim.x + threadIdx.x; i < n;
       i += (size_t)gridDim.x * blockDim.x)
    p[i] = 0.f;
}

__global__ __launch_bounds__(256)
void attn_div_kernel(float* __restrict__ wV, const float* __restrict__ Z)
{
  const size_t n = (size_t)NN * DM;
  for (size_t i = (size_t)blockIdx.x * blockDim.x + threadIdx.x; i < n;
       i += (size_t)gridDim.x * blockDim.x) {
    int c = (int)(i & 127);
    size_t node = i >> 7;
    wV[i] = wV[i] / (Z[node * 8 + (c >> 4)] + 1e-6f);
  }
}

// column-wise sum / sum-of-squares over [nrows x 128]
__global__ __launch_bounds__(256)
void bn_colstats_kernel(const float* __restrict__ X, float* __restrict__ sums,
                        int nrows)
{
  int c  = threadIdx.x & 127;
  int rl = threadIdx.x >> 7;     // 0..1
  float s = 0.f, s2 = 0.f;
  for (int r = blockIdx.x * 2 + rl; r < nrows; r += gridDim.x * 2) {
    float v = X[(size_t)r * DM + c];
    s += v; s2 += v * v;
  }
  atomicAdd(&sums[c], s);
  atomicAdd(&sums[DM + c], s2);
}

__global__ void bn_finalize_kernel(const float* __restrict__ sums,
                                   float* __restrict__ stats, float invn)
{
  int c = threadIdx.x;           // 128 threads
  float mu  = sums[c] * invn;
  float var = sums[DM + c] * invn - mu * mu;   // population var (jnp.var)
  stats[c]      = mu;
  stats[DM + c] = rsqrtf(var + 1e-5f);
}

__global__ __launch_bounds__(256)
void bn_apply_kernel(const float* __restrict__ X, const float* __restrict__ stats,
                     const float* __restrict__ g, const float* __restrict__ b,
                     float* __restrict__ Y)
{
  const size_t n = (size_t)NN * DM;
  for (size_t i = (size_t)blockIdx.x * blockDim.x + threadIdx.x; i < n;
       i += (size_t)gridDim.x * blockDim.x) {
    int c = (int)(i & 127);
    Y[i] = (X[i] - stats[c]) * stats[DM + c] * g[c] + b[c];
  }
}

// ---------------------------------------------------------------------------
extern "C" void kernel_launch(void* const* d_in, const int* in_sizes, int n_in,
                              void* d_out, int out_size, void* d_ws, size_t ws_size,
                              hipStream_t stream) {
  (void)in_sizes; (void)n_in; (void)out_size; (void)ws_size;

  const float* x         = (const float*)d_in[0];
  const float* edge_attr = (const float*)d_in[1];
  const int*   edge_idx  = (const int*)d_in[2];
  const float* Wq  = (const float*)d_in[3];
  const float* Wk  = (const float*)d_in[4];
  const float* Wep = (const float*)d_in[5];
  const float* Wv  = (const float*)d_in[6];
  const float* Wo  = (const float*)d_in[7];
  const float* bo  = (const float*)d_in[8];
  const float* g1  = (const float*)d_in[9];
  const float* b1  = (const float*)d_in[10];
  const float* W1  = (const float*)d_in[11];
  const float* bf1 = (const float*)d_in[12];
  const float* W2  = (const float*)d_in[13];
  const float* bf2 = (const float*)d_in[14];
  const float* g2  = (const float*)d_in[15];
  const float* b2  = (const float*)d_in[16];
  float* out = (float*)d_out;

  const size_t NMf = (size_t)NN * DM;       // node-matrix floats
  float* Qb   = (float*)d_ws;               // [N,128]  Q, later pre-BN1
  float* Kb   = Qb + NMf;                   // [N,128]  K, later h1 (h_in2)
  float* Vb   = Kb + NMf;                   // [N,128]  V
  float* wV   = Vb + NMf;                   // [N,128]  segment sums
  float* Zb   = wV + NMf;                   // [N,8]
  float* sums = Zb + (size_t)NN * 8;        // [256]
  float* stats = sums + 256;                // [256] mu | rsigma
  float* ff   = Vb;                         // [N,256] aliases dead V+wV span

  const dim3 blk256(256);
  const dim3 gN128((NN + 127) / 128, 1);    // 391 x 1
  const dim3 gN256((NN + 127) / 128, 2);    // 391 x 2

  // --- Q/K/V projections (WMMA) ---
  gemm_wmma_kernel<<<gN128, blk256, 0, stream>>>(x, Wq, Qb, nullptr, nullptr, NN, 128, 128, 0);
  gemm_wmma_kernel<<<gN128, blk256, 0, stream>>>(x, Wk, Kb, nullptr, nullptr, NN, 128, 128, 0);
  gemm_wmma_kernel<<<gN128, blk256, 0, stream>>>(x, Wv, Vb, nullptr, nullptr, NN, 128, 128, 0);

  // --- edge scatter phase ---
  zero_kernel<<<2048, blk256, 0, stream>>>(wV, NMf + (size_t)NN * 8); // wV + Z contiguous
  edge_attn_kernel<<<4096, dim3(128), 0, stream>>>(Qb, Kb, Vb, edge_attr,
                                                   edge_idx, Wep, wV, Zb);

  // --- h_attn = wV / (Z + 1e-6), then @Wo + bo + x (residual 1) -> Qb ---
  attn_div_kernel<<<4096, blk256, 0, stream>>>(wV, Zb);
  gemm_wmma_kernel<<<gN128, blk256, 0, stream>>>(wV, Wo, Qb, bo, x, NN, 128, 128, 0);

  // --- BatchNorm 1 -> h1 in Kb ---
  zero_kernel<<<1, blk256, 0, stream>>>(sums, 256);
  bn_colstats_kernel<<<512, blk256, 0, stream>>>(Qb, sums, NN);
  bn_finalize_kernel<<<1, dim3(128), 0, stream>>>(sums, stats, 1.0f / (float)NN);
  bn_apply_kernel<<<4096, blk256, 0, stream>>>(Qb, stats, g1, b1, Kb);

  // --- FFN: relu(h1@W1+bf1) -> ff ; ff@W2+bf2 + h1 -> out ---
  gemm_wmma_kernel<<<gN256, blk256, 0, stream>>>(Kb, W1, ff, bf1, nullptr, NN, 128, 256, 1);
  gemm_wmma_kernel<<<gN128, blk256, 0, stream>>>(ff, W2, out, bf2, Kb, NN, 256, 128, 0);

  // --- BatchNorm 2 in place on out ---
  zero_kernel<<<1, blk256, 0, stream>>>(sums, 256);
  bn_colstats_kernel<<<512, blk256, 0, stream>>>(out, sums, NN);
  bn_finalize_kernel<<<1, dim3(128), 0, stream>>>(sums, stats, 1.0f / (float)NN);
  bn_apply_kernel<<<4096, blk256, 0, stream>>>(out, stats, g2, b2, out);
}
